// BasicBlock_77137612636387
// MI455X (gfx1250) — matrix operand order repaired and checked by reading, not compiled, not run on hardware
//
#include <hip/hip_runtime.h>

// ---------------------------------------------------------------------------
// Problem constants (match reference)
// ---------------------------------------------------------------------------
#define B_SZ   1024
#define L_SZ   168
#define F_SZ   8
#define H_SZ   1024
#define KD2    3072      // H * K(=3) : reduction dim of conv2-tail GEMM
#define T_FORE 24
#define T_EST  168
#define LF     (L_SZ * F_SZ)   // 1344
#define AROW   40        // LDS A-tile row stride in bf16 elems (80B, bank-conflict-free)

typedef __bf16 bf16_t;
typedef __bf16 v16bf __attribute__((ext_vector_type(16)));
typedef __bf16 v8bf  __attribute__((ext_vector_type(8)));
typedef float  v8f   __attribute__((ext_vector_type(8)));

__device__ __forceinline__ float sigmoidf_(float x) {
    return 1.0f / (1.0f + __expf(-x));
}

// ---------------------------------------------------------------------------
// f32 -> bf16 weight conversion
// ---------------------------------------------------------------------------
__global__ void k_f32_to_bf16(const float* __restrict__ src,
                              bf16_t* __restrict__ dst, int n) {
    int i = blockIdx.x * 256 + threadIdx.x;
    if (i < n) dst[i] = (bf16_t)src[i];
}

// ---------------------------------------------------------------------------
// Conv1 tail: out1[b, c, L-3..L-1] = relu(causal conv1(x)) ; stores A2 bf16
// A2[b, c*3 + k]  (K-layout matches conv2_w flat [h][c][k])
// ---------------------------------------------------------------------------
__global__ void k_conv1_tail(const float* __restrict__ inp,
                             const float* __restrict__ w1,
                             const float* __restrict__ b1,
                             bf16_t* __restrict__ A2) {
    __shared__ float xs[40];                 // [t-163][f]
    const int b   = blockIdx.y;
    const int tid = threadIdx.x;
    if (tid < 40) {
        int t = 163 + (tid >> 3), f = tid & 7;
        xs[tid] = inp[b * LF + t * F_SZ + f];
    }
    __syncthreads();
    const int c = blockIdx.x * 256 + tid;    // grid.x = 4 -> c in [0,1024)
    float w[24];
#pragma unroll
    for (int i = 0; i < 24; ++i) w[i] = w1[c * 24 + i];
    const float bias = b1[c];
#pragma unroll
    for (int k = 0; k < 3; ++k) {            // output t = 165 + k
        float acc = bias;
#pragma unroll
        for (int kk = 0; kk < 3; ++kk)
#pragma unroll
            for (int f = 0; f < 8; ++f)
                acc += xs[(k + kk) * 8 + f] * w[f * 3 + kk];
        A2[(size_t)b * KD2 + c * 3 + k] = (bf16_t)fmaxf(acc, 0.0f);
    }
}

// ---------------------------------------------------------------------------
// Conv2-tail GEMM + residual + relu -> h0 (f32 for GRU epilogue, bf16 for WMMA)
// Block: 128(b) x 64(j), 8 waves = 4(m) x 2(n); wave computes 32x32 (2x2 WMMA).
// ---------------------------------------------------------------------------
__global__ __launch_bounds__(256)
void k_conv2_gemm(const bf16_t* __restrict__ A2, const bf16_t* __restrict__ W2,
                  const float* __restrict__ b2,  const float* __restrict__ dw,
                  const float* __restrict__ db,  const float* __restrict__ inp,
                  float* __restrict__ h0f, bf16_t* __restrict__ h0bf) {
    const int lane = threadIdx.x & 31;
    const int wave = threadIdx.x >> 5;
    const int wm = wave & 3, wn = wave >> 2;
    const int b0 = blockIdx.y * 128 + wm * 32;
    const int j0 = blockIdx.x * 64  + wn * 32;
    const int n  = lane & 15, hi = lane >> 4;

    v8f acc[2][2] = {};                      // [u: n-tile][t: m-tile]

    for (int kc = 0; kc < KD2; kc += 32) {
        v16bf a[2], bb[2];
#pragma unroll
        for (int t = 0; t < 2; ++t) {
            const bf16_t* base = A2 + (size_t)(b0 + 16 * t + n) * KD2 + kc + hi * 8;
            v8bf lo = *(const v8bf*)(base);
            v8bf up = *(const v8bf*)(base + 16);
#pragma unroll
            for (int i = 0; i < 8; ++i) { a[t][i] = lo[i]; a[t][i + 8] = up[i]; }
        }
#pragma unroll
        for (int u = 0; u < 2; ++u) {
            const bf16_t* wb = W2 + (size_t)(j0 + 16 * u + n) * KD2 + kc + hi * 8;
            v8bf lo = *(const v8bf*)(wb);
            v8bf up = *(const v8bf*)(wb + 16);
#pragma unroll
            for (int i = 0; i < 8; ++i) { bb[u][i] = lo[i]; bb[u][i + 8] = up[i]; }
        }
#pragma unroll
        for (int u = 0; u < 2; ++u)
#pragma unroll
            for (int t = 0; t < 2; ++t)
                acc[u][t] = __builtin_amdgcn_wmma_f32_16x16x32_bf16(
                    false, a[t], false, bb[u], (short)0, acc[u][t], false, false);
    }

    float biasu[2], w8[2][8];
#pragma unroll
    for (int u = 0; u < 2; ++u) {
        const int j = j0 + 16 * u + n;
        biasu[u] = b2[j] + db[j];
#pragma unroll
        for (int f = 0; f < 8; ++f) w8[u][f] = dw[j * 8 + f];
    }
#pragma unroll
    for (int t = 0; t < 2; ++t)
#pragma unroll
        for (int v = 0; v < 8; ++v) {
            const int b = b0 + 16 * t + v + 8 * hi;
            float x8[8];
#pragma unroll
            for (int f = 0; f < 8; ++f) x8[f] = inp[b * LF + 167 * F_SZ + f];
#pragma unroll
            for (int u = 0; u < 2; ++u) {
                const int j = j0 + 16 * u + n;
                float r = biasu[u];
#pragma unroll
                for (int f = 0; f < 8; ++f) r += x8[f] * w8[u][f];
                const float hv = fmaxf(acc[u][t][v] + r, 0.0f);
                h0f[(size_t)b * H_SZ + j]  = hv;
                h0bf[(size_t)b * H_SZ + j] = (bf16_t)hv;
            }
        }
}

// ---------------------------------------------------------------------------
// Fused GRU step. A-tile (h, bf16) staged to LDS with gfx1250 async-to-LDS,
// double-buffered on ASYNCcnt; weights streamed from L2 (6MB, resident).
// Block: 128(b) x 32(j), 8 waves = 4(m) x 2(n); per wave 32b x 16j x 3 gates
// -> 6 v_wmma per K-chunk, 32 chunks. gridDim.z selects decoder (fused pair).
// ---------------------------------------------------------------------------
struct GruArgs {
    const bf16_t* hbf_in;
    const float*  hf_in;
    bf16_t*       hbf_out;
    float*        hf_out;
    const bf16_t* Whh;
    const float*  w_ih;
    const float*  b_ih;
    const float*  b_hh;
    const float*  fc_w;
    const float*  x_in;
    float*        pred_out;
};

__global__ __launch_bounds__(256)
void k_gru_step(GruArgs A0, GruArgs A1) {
    const GruArgs P = (blockIdx.z == 0) ? A0 : A1;
    __shared__ __align__(32) bf16_t abuf[2][128 * AROW];

    const int lane = threadIdx.x & 31;
    const int wave = threadIdx.x >> 5;
    const int wm = wave & 3, wn = wave >> 2;
    const int bblk = blockIdx.y * 128;
    const int j0 = blockIdx.x * 32 + wn * 16;
    const int n  = lane & 15, hi = lane >> 4;

    // --- async A-tile copy assignment: wave w copies rows [16w,16w+16),
    //     2 issues of 8 rows; lane -> (row = q*8 + lane/4, 16B segment = lane&3)
    const int crow = wave * 16 + (lane >> 2);
    const int cseg = lane & 3;
    const uint32_t lbase =
        (uint32_t)(uintptr_t)(&abuf[0][0]) + (uint32_t)((crow * AROW + cseg * 8) * 2);
    const uint64_t gbase =
        (uint64_t)(uintptr_t)(P.hbf_in + (size_t)(bblk + crow) * H_SZ + cseg * 8);
    const uint32_t lbufstride = (uint32_t)(128 * AROW * 2);

    v8f acc[3][2] = {};                      // [gate][m-tile]
    const int NCH = H_SZ / 32;               // 32 K-chunks

    // prologue: issue chunk 0 into buf 0
#pragma unroll
    for (int q = 0; q < 2; ++q) {
        uint32_t l = lbase + (uint32_t)(q * 8 * AROW * 2);
        uint64_t g = gbase + (uint64_t)(q * 8 * H_SZ * 2);
        asm volatile("global_load_async_to_lds_b128 %0, %1, off"
                     :: "v"(l), "v"(g) : "memory");
    }

    for (int i = 0; i < NCH; ++i) {
        if (i + 1 < NCH) {
            // prefetch next chunk into the other buffer
#pragma unroll
            for (int q = 0; q < 2; ++q) {
                uint32_t l = lbase + (uint32_t)(((i + 1) & 1)) * lbufstride
                           + (uint32_t)(q * 8 * AROW * 2);
                uint64_t g = gbase + (uint64_t)((i + 1) * 32 * 2)
                           + (uint64_t)(q * 8 * H_SZ * 2);
                asm volatile("global_load_async_to_lds_b128 %0, %1, off"
                             :: "v"(l), "v"(g) : "memory");
            }
            asm volatile("s_wait_asynccnt 2" ::: "memory");   // chunk i landed
        } else {
            asm volatile("s_wait_asynccnt 0" ::: "memory");
        }
        __syncthreads();                                      // visible to all waves

        // A fragments from LDS (conflict-free: 80B row stride)
        const bf16_t* ab = &abuf[i & 1][0];
        v16bf a[2];
#pragma unroll
        for (int t = 0; t < 2; ++t) {
            const bf16_t* r = ab + (wm * 32 + 16 * t + n) * AROW + hi * 8;
            v8bf lo = *(const v8bf*)(r);
            v8bf up = *(const v8bf*)(r + 16);
#pragma unroll
            for (int ii = 0; ii < 8; ++ii) { a[t][ii] = lo[ii]; a[t][ii + 8] = up[ii]; }
        }
        const int kc = i * 32;
#pragma unroll
        for (int g = 0; g < 3; ++g) {
            const bf16_t* wb = P.Whh + ((size_t)g * H_SZ + j0 + n) * H_SZ
                                       + kc + hi * 8;
            v8bf blo = *(const v8bf*)(wb);
            v8bf bup = *(const v8bf*)(wb + 16);
            v16bf bb;
#pragma unroll
            for (int ii = 0; ii < 8; ++ii) { bb[ii] = blo[ii]; bb[ii + 8] = bup[ii]; }
#pragma unroll
            for (int t = 0; t < 2; ++t)
                acc[g][t] = __builtin_amdgcn_wmma_f32_16x16x32_bf16(
                    false, a[t], false, bb, (short)0, acc[g][t], false, false);
        }
        __syncthreads();    // done reading buf before it is refilled (i+2)
    }

    // ---- epilogue: gates, state update, fc projection ----
    const int j = j0 + n;
    const float wir = P.w_ih[j], wiz = P.w_ih[H_SZ + j], win = P.w_ih[2 * H_SZ + j];
    const float bir = P.b_ih[j], biz = P.b_ih[H_SZ + j], bin = P.b_ih[2 * H_SZ + j];
    const float bhr = P.b_hh[j], bhz = P.b_hh[H_SZ + j], bhn = P.b_hh[2 * H_SZ + j];
    const float fcw = P.fc_w[j];
    const int b0 = bblk + wm * 32;

#pragma unroll
    for (int t = 0; t < 2; ++t)
#pragma unroll
        for (int v = 0; v < 8; ++v) {
            const int b = b0 + 16 * t + v + 8 * hi;
            const float x = P.x_in[b];
            const float rg = sigmoidf_(x * wir + bir + acc[0][t][v] + bhr);
            const float zg = sigmoidf_(x * wiz + biz + acc[1][t][v] + bhz);
            const float ng = tanhf(x * win + bin + rg * (acc[2][t][v] + bhn));
            const float hp = P.hf_in[(size_t)b * H_SZ + j];
            const float hn = (1.0f - zg) * ng + zg * hp;
            P.hf_out[(size_t)b * H_SZ + j]  = hn;
            P.hbf_out[(size_t)b * H_SZ + j] = (bf16_t)hn;
            float part = hn * fcw;            // fc partial over this wave's 16 j's
            part += __shfl_xor(part, 1);
            part += __shfl_xor(part, 2);
            part += __shfl_xor(part, 4);
            part += __shfl_xor(part, 8);
            if (n == 0) atomicAdd(&P.pred_out[b], part);
        }
}

// ---------------------------------------------------------------------------
// Init: pred accumulators <- fc_b ; x0 seeds
// ---------------------------------------------------------------------------
__global__ void k_init(float* __restrict__ predF, float* __restrict__ predE,
                       float* __restrict__ x0F, float* __restrict__ x0E,
                       const float* __restrict__ inp,
                       const float* __restrict__ ffcb, const float* __restrict__ efcb) {
    int i = blockIdx.x * 256 + threadIdx.x;
    if (i < T_FORE * B_SZ) predF[i] = ffcb[0];
    if (i < T_EST * B_SZ)  predE[i] = efcb[0];
    if (i < B_SZ) {
        x0F[i] = inp[i * LF + 167 * F_SZ];   // x[:,0,-1]
        x0E[i] = inp[i * LF];                // x[:,0,0]
    }
}

// ---------------------------------------------------------------------------
// Final: pack [B,24,1] forecast then [B,168,1] (input_main - estimate)
// ---------------------------------------------------------------------------
__global__ void k_final(const float* __restrict__ predF, const float* __restrict__ predE,
                        const float* __restrict__ inp, float* __restrict__ out) {
    const int i  = blockIdx.x * 256 + threadIdx.x;
    const int NF = T_FORE * B_SZ;
    if (i < NF) {
        int b = i / T_FORE, s = i % T_FORE;
        out[i] = predF[s * B_SZ + b];
    } else if (i < NF + T_EST * B_SZ) {
        int k = i - NF;
        int b = k / T_EST, s = k % T_EST;
        out[i] = inp[b * LF + s * F_SZ] - predE[s * B_SZ + b];
    }
}

// ---------------------------------------------------------------------------
extern "C" void kernel_launch(void* const* d_in, const int* in_sizes, int n_in,
                              void* d_out, int out_size, void* d_ws, size_t ws_size,
                              hipStream_t stream) {
    (void)in_sizes; (void)n_in; (void)out_size; (void)ws_size;
    const float* inp     = (const float*)d_in[0];
    const float* conv1_w = (const float*)d_in[1];
    const float* conv1_b = (const float*)d_in[2];
    const float* conv2_w = (const float*)d_in[3];
    const float* conv2_b = (const float*)d_in[4];
    const float* down_w  = (const float*)d_in[5];
    const float* down_b  = (const float*)d_in[6];
    const float* f_w_ih  = (const float*)d_in[7];
    const float* f_w_hh  = (const float*)d_in[8];
    const float* f_b_ih  = (const float*)d_in[9];
    const float* f_b_hh  = (const float*)d_in[10];
    const float* f_fc_w  = (const float*)d_in[11];
    const float* f_fc_b  = (const float*)d_in[12];
    const float* e_w_ih  = (const float*)d_in[13];
    const float* e_w_hh  = (const float*)d_in[14];
    const float* e_b_ih  = (const float*)d_in[15];
    const float* e_b_hh  = (const float*)d_in[16];
    const float* e_fc_w  = (const float*)d_in[17];
    const float* e_fc_b  = (const float*)d_in[18];

    // ---- workspace carve-up (256B aligned slabs) ----
    char* ws = (char*)d_ws;
    size_t off = 0;
    auto carve = [&](size_t bytes) -> void* {
        void* p = ws + off;
        off += (bytes + 255) & ~(size_t)255;
        return p;
    };
    const size_t NW = (size_t)3 * H_SZ * H_SZ;            // 3,145,728 weights
    const size_t NH = (size_t)B_SZ * H_SZ;
    bf16_t* W2bf = (bf16_t*)carve(NW * 2);                // conv2_w bf16 [H][3072]
    bf16_t* WhhF = (bf16_t*)carve(NW * 2);                // f_w_hh bf16 [3H][H]
    bf16_t* WhhE = (bf16_t*)carve(NW * 2);                // e_w_hh bf16 [3H][H]
    bf16_t* A2   = (bf16_t*)carve((size_t)B_SZ * KD2 * 2);
    float*  h0f  = (float*)carve(NH * 4);
    bf16_t* h0bf = (bf16_t*)carve(NH * 2);
    float*  hFf[2]  = {(float*)carve(NH * 4), (float*)carve(NH * 4)};
    float*  hEf[2]  = {(float*)carve(NH * 4), (float*)carve(NH * 4)};
    bf16_t* hFbf[2] = {(bf16_t*)carve(NH * 2), (bf16_t*)carve(NH * 2)};
    bf16_t* hEbf[2] = {(bf16_t*)carve(NH * 2), (bf16_t*)carve(NH * 2)};
    float*  predF = (float*)carve((size_t)T_FORE * B_SZ * 4);
    float*  predE = (float*)carve((size_t)T_EST * B_SZ * 4);
    float*  x0F   = (float*)carve(B_SZ * 4);
    float*  x0E   = (float*)carve(B_SZ * 4);

    // ---- weight conversion ----
    const int cvtBlocks = (int)((NW + 255) / 256);
    k_f32_to_bf16<<<cvtBlocks, 256, 0, stream>>>(conv2_w, W2bf, (int)NW);
    k_f32_to_bf16<<<cvtBlocks, 256, 0, stream>>>(f_w_hh, WhhF, (int)NW);
    k_f32_to_bf16<<<cvtBlocks, 256, 0, stream>>>(e_w_hh, WhhE, (int)NW);

    // ---- init pred accumulators + x0 ----
    k_init<<<(T_EST * B_SZ + 255) / 256, 256, 0, stream>>>(
        predF, predE, x0F, x0E, inp, f_fc_b, e_fc_b);

    // ---- encoder (last-timestep receptive field only) ----
    k_conv1_tail<<<dim3(H_SZ / 256, B_SZ), 256, 0, stream>>>(inp, conv1_w, conv1_b, A2);
    k_conv2_gemm<<<dim3(H_SZ / 64, B_SZ / 128), 256, 0, stream>>>(
        A2, W2bf, conv2_b, down_w, down_b, inp, h0f, h0bf);

    // ---- decoders: estimate 168 steps; forecast fused via gridDim.z for s<24 ----
    for (int s = 0; s < T_EST; ++s) {
        GruArgs aE;
        aE.hbf_in  = (s == 0) ? h0bf : hEbf[(s + 1) & 1];
        aE.hf_in   = (s == 0) ? h0f  : hEf[(s + 1) & 1];
        aE.hbf_out = hEbf[s & 1];
        aE.hf_out  = hEf[s & 1];
        aE.Whh = WhhE; aE.w_ih = e_w_ih; aE.b_ih = e_b_ih; aE.b_hh = e_b_hh;
        aE.fc_w = e_fc_w;
        aE.x_in = (s == 0) ? x0E : predE + (size_t)(s - 1) * B_SZ;
        aE.pred_out = predE + (size_t)s * B_SZ;

        if (s < T_FORE) {
            GruArgs aF;
            aF.hbf_in  = (s == 0) ? h0bf : hFbf[(s + 1) & 1];
            aF.hf_in   = (s == 0) ? h0f  : hFf[(s + 1) & 1];
            aF.hbf_out = hFbf[s & 1];
            aF.hf_out  = hFf[s & 1];
            aF.Whh = WhhF; aF.w_ih = f_w_ih; aF.b_ih = f_b_ih; aF.b_hh = f_b_hh;
            aF.fc_w = f_fc_w;
            aF.x_in = (s == 0) ? x0F : predF + (size_t)(s - 1) * B_SZ;
            aF.pred_out = predF + (size_t)s * B_SZ;
            k_gru_step<<<dim3(H_SZ / 32, B_SZ / 128, 2), 256, 0, stream>>>(aF, aE);
        } else {
            k_gru_step<<<dim3(H_SZ / 32, B_SZ / 128, 1), 256, 0, stream>>>(aE, aE);
        }
    }

    // ---- pack outputs ----
    const int NOUT = T_FORE * B_SZ + T_EST * B_SZ;
    k_final<<<(NOUT + 255) / 256, 256, 0, stream>>>(predF, predE, inp, (float*)d_out);
}
